// MLPBlock_81819126989555
// MI455X (gfx1250) — matrix operand order repaired and checked by reading, not compiled, not run on hardware
//
#include <hip/hip_runtime.h>

// Problem constants (from reference)
#define T_TOK  4096
#define H_DIM  1024
#define E_NUM  8
#define I_DIM  2048
#define TWO_I  4096
#define TOPK   2
#define EPSV   1e-5f

typedef __attribute__((ext_vector_type(16))) __bf16 v16bf;
typedef __attribute__((ext_vector_type(8)))  __bf16 v8bf;
typedef __attribute__((ext_vector_type(8)))  float  v8f;
typedef int b128_t __attribute__((vector_size(16)));   // matches builtin param

union BF16Frag { v16bf v; v8bf h[2]; };

#define AS1 __attribute__((address_space(1)))
#define AS3 __attribute__((address_space(3)))

// ---------------------------------------------------------------------------
// Async global->LDS copy of one 16B chunk (CDNA5 GLOBAL_LOAD_ASYNC_TO_LDS_B128,
// tracked by ASYNCcnt). Falls back to a synchronous vector copy if the builtin
// is unavailable on this toolchain.
// ---------------------------------------------------------------------------
#if __has_builtin(__builtin_amdgcn_global_load_async_to_lds_b128)
#define HAVE_ASYNC_LDS 1
#else
#define HAVE_ASYNC_LDS 0
#endif

__device__ __forceinline__ void async_copy16(const __bf16* g, __bf16* l) {
#if HAVE_ASYNC_LDS
  __builtin_amdgcn_global_load_async_to_lds_b128(
      (AS1 b128_t*)(b128_t*)(__bf16*)g,   // drop const, retype, -> addrspace(1)
      (AS3 b128_t*)(b128_t*)l,            // retype, -> addrspace(3)
      0, 0);
#else
  *(v8bf*)l = *(const v8bf*)g;
#endif
}

template <int N>
__device__ __forceinline__ void wait_async_le() {
#if HAVE_ASYNC_LDS
#if __has_builtin(__builtin_amdgcn_s_wait_asynccnt)
  __builtin_amdgcn_s_wait_asynccnt(N);
#else
  asm volatile("s_wait_asynccnt %0" :: "i"(N) : "memory");
#endif
#endif
}

// ---------------------------------------------------------------------------
// Kernel 0: tiled transpose + f32->bf16 convert.  in: [z][R][C] f32,
// out: [z][C][R] bf16.  Makes every GEMM B-tile a contiguous row block so it
// can be staged with async b128 copies.
// ---------------------------------------------------------------------------
__global__ __launch_bounds__(256) void transpose_cvt_kernel(
    const float* __restrict__ in, __bf16* __restrict__ out, int R, int C)
{
  __shared__ float tile[32][33];
  const int bx = blockIdx.x * 32;                 // C coordinate
  const int by = blockIdx.y * 32;                 // R coordinate
  const size_t base = (size_t)blockIdx.z * R * C;
  const int tx = threadIdx.x & 31, ty = threadIdx.x >> 5;   // 32 x 8
#pragma unroll
  for (int r = ty; r < 32; r += 8)
    tile[r][tx] = in[base + (size_t)(by + r) * C + (bx + tx)];
  __syncthreads();
#pragma unroll
  for (int c = ty; c < 32; c += 8)
    out[base + (size_t)(bx + c) * R + (by + tx)] = (__bf16)tile[tx][c];
}

// ---------------------------------------------------------------------------
// Kernel 1: RMSNorm -> bf16
// ---------------------------------------------------------------------------
__global__ __launch_bounds__(256) void rmsnorm_kernel(
    const float* __restrict__ x, const float* __restrict__ scale,
    __bf16* __restrict__ normed)
{
  const int t   = blockIdx.x;
  const int tid = threadIdx.x;
  __shared__ float red[256];

  const float* row = x + (size_t)t * H_DIM;
  float s = 0.f;
  for (int i = tid; i < H_DIM; i += 256) { float v = row[i]; s += v * v; }
  red[tid] = s;
  __syncthreads();
  for (int off = 128; off > 0; off >>= 1) {
    if (tid < off) red[tid] += red[tid + off];
    __syncthreads();
  }
  const float inv = rsqrtf(red[0] * (1.0f / H_DIM) + EPSV);
  __bf16* orow = normed + (size_t)t * H_DIM;
  for (int i = tid; i < H_DIM; i += 256)
    orow[i] = (__bf16)(row[i] * inv * scale[i]);
}

// ---------------------------------------------------------------------------
// Kernel 2: hidden = normed @ W1[e] (+b1), fused clip/sigmoid activation.
// Async double-buffered LDS staging; 8 waves, wave tile 32x32 (2x2 WMMA),
// gate + linear column tiles computed together so activation fuses.
// ---------------------------------------------------------------------------
__global__ __launch_bounds__(256) void mlp1_act_kernel(
    const __bf16* __restrict__ normed,   // [T, H] bf16
    const __bf16* __restrict__ w1t,      // [TOPK, 2I, H] bf16 (transposed)
    const float*  __restrict__ b1,       // [E, 2I] f32
    __bf16*       __restrict__ act)      // [TOPK, T, I] bf16
{
  const int e    = blockIdx.z;
  const int n0   = blockIdx.x * 64;      // column tile within I
  const int m0   = blockIdx.y * 128;     // token tile
  const int tid  = threadIdx.x;
  const int lane = tid & 31;
  const int wave = tid >> 5;
  const int wm   = wave & 3;             // 0..3 (M)
  const int wn   = wave >> 2;            // 0..1 (N)

  __shared__ __align__(16) __bf16 lA [2][128][32];
  __shared__ __align__(16) __bf16 lBg[2][64][32];
  __shared__ __align__(16) __bf16 lBl[2][64][32];

  const __bf16* w1te = w1t + (size_t)e * TWO_I * H_DIM;

  const v8f zacc = {};
  v8f accg[2][2], accl[2][2];
#pragma unroll
  for (int i = 0; i < 2; ++i)
#pragma unroll
    for (int j = 0; j < 2; ++j) { accg[i][j] = zacc; accl[i][j] = zacc; }

  const int rowA = tid >> 1, hv = tid & 1;   // A staging coords
  const int nB   = tid >> 2, q  = tid & 3;   // B staging coords

  auto stage = [&](int buf, int k0) {
    const __bf16* gA = normed + (size_t)(m0 + rowA) * H_DIM + k0 + hv * 16;
    async_copy16(gA,     &lA[buf][rowA][hv * 16]);
    async_copy16(gA + 8, &lA[buf][rowA][hv * 16 + 8]);
    async_copy16(w1te + (size_t)(n0 + nB) * H_DIM + k0 + q * 8,
                 &lBg[buf][nB][q * 8]);
    async_copy16(w1te + (size_t)(I_DIM + n0 + nB) * H_DIM + k0 + q * 8,
                 &lBl[buf][nB][q * 8]);
  };

  const int NK = H_DIM / 32;
  stage(0, 0);
  for (int kk = 0; kk < NK; ++kk) {
    const int cur = kk & 1;
    if ((kk + 1) < NK) {           // wave-uniform branch
      stage(cur ^ 1, (kk + 1) * 32);
      wait_async_le<4>();          // in-order: tile `cur` has landed
    } else {
      wait_async_le<0>();
    }
    __syncthreads();

    const int mrow = lane & 15, klo = (lane >> 4) * 16;
    const int ncol = lane & 15, kb  = (lane >> 4) * 16;
    BF16Frag a[2], bg[2], bl[2];
#pragma unroll
    for (int tm = 0; tm < 2; ++tm) {
      const int m = wm * 32 + tm * 16 + mrow;
      a[tm].h[0] = *(const v8bf*)&lA[cur][m][klo];
      a[tm].h[1] = *(const v8bf*)&lA[cur][m][klo + 8];
    }
#pragma unroll
    for (int tn = 0; tn < 2; ++tn) {
      const int n = wn * 32 + tn * 16 + ncol;
      bg[tn].h[0] = *(const v8bf*)&lBg[cur][n][kb];
      bg[tn].h[1] = *(const v8bf*)&lBg[cur][n][kb + 8];
      bl[tn].h[0] = *(const v8bf*)&lBl[cur][n][kb];
      bl[tn].h[1] = *(const v8bf*)&lBl[cur][n][kb + 8];
    }
#pragma unroll
    for (int tm = 0; tm < 2; ++tm)
#pragma unroll
      for (int tn = 0; tn < 2; ++tn) {
        accg[tm][tn] = __builtin_amdgcn_wmma_f32_16x16x32_bf16(
            false, a[tm].v, false, bg[tn].v, (short)0, accg[tm][tn], false, false);
        accl[tm][tn] = __builtin_amdgcn_wmma_f32_16x16x32_bf16(
            false, a[tm].v, false, bl[tn].v, (short)0, accl[tm][tn], false, false);
      }
    __syncthreads();   // protect buf `cur` before it is re-staged next iter
  }

  // epilogue: bias + clip + gate*sigmoid(1.702*gate)*(linear+1) -> bf16
  const float* b1e  = b1 + (size_t)e * TWO_I;
  __bf16*      acte = act + (size_t)e * T_TOK * I_DIM;
#pragma unroll
  for (int tm = 0; tm < 2; ++tm)
#pragma unroll
    for (int tn = 0; tn < 2; ++tn)
#pragma unroll
      for (int r = 0; r < 8; ++r) {
        const int m  = m0 + wm * 32 + tm * 16 + r + ((lane >> 4) << 3);
        const int nI = n0 + wn * 32 + tn * 16 + (lane & 15);
        const float hg = accg[tm][tn][r] + b1e[nI];
        const float hl = accl[tm][tn][r] + b1e[I_DIM + nI];
        const float g  = fminf(hg, 7.f);
        const float li = fminf(fmaxf(hl, -7.f), 7.f);
        const float av = g / (1.f + __expf(-1.702f * g)) * (li + 1.f);
        acte[(size_t)m * I_DIM + nI] = (__bf16)av;
      }
}

// ---------------------------------------------------------------------------
// Kernel 3: out = x + 0.5*sum_e (act_e @ W2[e]) + 0.5*sum_e b2[e]
// ---------------------------------------------------------------------------
__global__ __launch_bounds__(256) void mlp2_mix_kernel(
    const __bf16* __restrict__ act,  // [TOPK, T, I] bf16
    const __bf16* __restrict__ w2t,  // [TOPK, H, I] bf16 (transposed)
    const float*  __restrict__ b2,   // [E, H] f32
    const float*  __restrict__ x,    // [T, H] f32
    float*        __restrict__ out)  // [T, H] f32
{
  const int n0   = blockIdx.x * 64;   // H tile
  const int m0   = blockIdx.y * 128;  // token tile
  const int tid  = threadIdx.x;
  const int lane = tid & 31;
  const int wave = tid >> 5;
  const int wm   = wave & 3;
  const int wn   = wave >> 2;

  __shared__ __align__(16) __bf16 lA[2][128][32];
  __shared__ __align__(16) __bf16 lB[2][64][32];

  const v8f zacc = {};
  v8f acc[2][2];
#pragma unroll
  for (int i = 0; i < 2; ++i)
#pragma unroll
    for (int j = 0; j < 2; ++j) acc[i][j] = zacc;

  const int rowA = tid >> 1, hv = tid & 1;
  const int nB   = tid >> 2, q  = tid & 3;

  // K runs over TOPK experts x I (concatenated)
  auto stage = [&](int buf, int kGlob) {
    const int e  = kGlob / I_DIM;
    const int k0 = kGlob % I_DIM;
    const __bf16* ae  = act + (size_t)e * T_TOK * I_DIM;
    const __bf16* we  = w2t + (size_t)e * H_DIM * I_DIM;
    const __bf16* gA  = ae + (size_t)(m0 + rowA) * I_DIM + k0 + hv * 16;
    async_copy16(gA,     &lA[buf][rowA][hv * 16]);
    async_copy16(gA + 8, &lA[buf][rowA][hv * 16 + 8]);
    async_copy16(we + (size_t)(n0 + nB) * I_DIM + k0 + q * 8,
                 &lB[buf][nB][q * 8]);
  };

  const int NK = (TOPK * I_DIM) / 32;
  stage(0, 0);
  for (int kk = 0; kk < NK; ++kk) {
    const int cur = kk & 1;
    if ((kk + 1) < NK) {           // wave-uniform branch
      stage(cur ^ 1, (kk + 1) * 32);
      wait_async_le<3>();
    } else {
      wait_async_le<0>();
    }
    __syncthreads();

    const int mrow = lane & 15, klo = (lane >> 4) * 16;
    const int ncol = lane & 15, kb  = (lane >> 4) * 16;
    BF16Frag a[2], b[2];
#pragma unroll
    for (int tm = 0; tm < 2; ++tm) {
      const int m = wm * 32 + tm * 16 + mrow;
      a[tm].h[0] = *(const v8bf*)&lA[cur][m][klo];
      a[tm].h[1] = *(const v8bf*)&lA[cur][m][klo + 8];
    }
#pragma unroll
    for (int tn = 0; tn < 2; ++tn) {
      const int n = wn * 32 + tn * 16 + ncol;
      b[tn].h[0] = *(const v8bf*)&lB[cur][n][kb];
      b[tn].h[1] = *(const v8bf*)&lB[cur][n][kb + 8];
    }
#pragma unroll
    for (int tm = 0; tm < 2; ++tm)
#pragma unroll
      for (int tn = 0; tn < 2; ++tn)
        acc[tm][tn] = __builtin_amdgcn_wmma_f32_16x16x32_bf16(
            false, a[tm].v, false, b[tn].v, (short)0, acc[tm][tn], false, false);
    __syncthreads();
  }

  // epilogue: residual + 0.5 mix + summed biases (experts 0,1)
#pragma unroll
  for (int tm = 0; tm < 2; ++tm)
#pragma unroll
    for (int tn = 0; tn < 2; ++tn)
#pragma unroll
      for (int r = 0; r < 8; ++r) {
        const int m = m0 + wm * 32 + tm * 16 + r + ((lane >> 4) << 3);
        const int h = n0 + wn * 32 + tn * 16 + (lane & 15);
        const float bias = 0.5f * (b2[h] + b2[H_DIM + h]);
        out[(size_t)m * H_DIM + h] =
            x[(size_t)m * H_DIM + h] + 0.5f * acc[tm][tn][r] + bias;
      }
}

// ---------------------------------------------------------------------------
// Launch
// ---------------------------------------------------------------------------
extern "C" void kernel_launch(void* const* d_in, const int* in_sizes, int n_in,
                              void* d_out, int out_size, void* d_ws, size_t ws_size,
                              hipStream_t stream) {
  const float* x     = (const float*)d_in[0];
  const float* scale = (const float*)d_in[1];
  // d_in[2] gate_kernel, d_in[3] gate_bias: dead (static routing, result unused)
  const float* w1 = (const float*)d_in[4];
  const float* b1 = (const float*)d_in[5];
  const float* w2 = (const float*)d_in[6];
  const float* b2 = (const float*)d_in[7];
  float* out = (float*)d_out;

  // Workspace carve (bf16): normed [T,H] | act [TOPK,T,I] | w1t [TOPK,2I,H]
  // | w2t [TOPK,H,I]  => ~67 MB total
  __bf16* normed = (__bf16*)d_ws;
  __bf16* actbuf = normed + (size_t)T_TOK * H_DIM;
  __bf16* w1t    = actbuf + (size_t)TOPK * T_TOK * I_DIM;
  __bf16* w2t    = w1t    + (size_t)TOPK * TWO_I * H_DIM;

  // Pre-transpose + convert live expert weights to bf16
  transpose_cvt_kernel<<<dim3(TWO_I / 32, H_DIM / 32, TOPK), dim3(256), 0, stream>>>(
      w1, w1t, H_DIM, TWO_I);
  transpose_cvt_kernel<<<dim3(H_DIM / 32, I_DIM / 32, TOPK), dim3(256), 0, stream>>>(
      w2, w2t, I_DIM, H_DIM);

  rmsnorm_kernel<<<dim3(T_TOK), dim3(256), 0, stream>>>(x, scale, normed);

  mlp1_act_kernel<<<dim3(I_DIM / 64, T_TOK / 128, TOPK), dim3(256), 0, stream>>>(
      normed, w1t, b1, actbuf);
  mlp2_mix_kernel<<<dim3(H_DIM / 64, T_TOK / 128), dim3(256), 0, stream>>>(
      actbuf, w2t, b2, x, out);
}
